// Model_67705864454830
// MI455X (gfx1250) — matrix operand order repaired and checked by reading, not compiled, not run on hardware
//
#include <hip/hip_runtime.h>
#include <hip/hip_bf16.h>

typedef __attribute__((ext_vector_type(16))) _Float16 v16h;
typedef __attribute__((ext_vector_type(8)))  _Float16 v8h;
typedef __attribute__((ext_vector_type(8)))  float    v8f;

#define GRID_N   128
#define CH       128
#define NB       2
#define NATOM    16
#define ALEN     8192
#define NSAMP    32768
#define LOWF     (30.0f / 11025.0f)

// ---------------------------------------------------------------------------
// Weight prepack: OIHW f32 [128,128,3,3] -> WMMA A-fragment layout f16
// layout: [tap 9][kblk 4][mtile 8][lane 32][half 16]
// lane<16 : M=lane,   K = {0..15}   (within 32-wide K block)
// lane>=16: M=lane-16,K = {16..31}
// ---------------------------------------------------------------------------
__global__ __launch_bounds__(256) void pack_weights_k(const float* __restrict__ w,
                                                      _Float16* __restrict__ out) {
    int i = blockIdx.x * 256 + threadIdx.x;
    if (i >= 9 * 4 * 8 * 32 * 16) return;
    int h    = i & 15;
    int lane = (i >> 4) & 31;
    int t    = (i >> 9) & 7;
    int kblk = (i >> 12) & 3;
    int tap  = i >> 14;
    int m      = t * 16 + (lane & 15);
    int hi     = lane >> 4;
    int klocal = hi * 16 + h;
    int inc    = kblk * 32 + klocal;
    int ky = tap / 3, kx = tap % 3;
    out[i] = (_Float16)w[((m * CH + inc) * 3 + ky) * 3 + kx];
}

// ---------------------------------------------------------------------------
// conv_in: x (B,1,128,128) * wi (112,1,3,3) + bi, concat pos (16,128,128)
// -> NHWC f16 (B,128,128,128)
// ---------------------------------------------------------------------------
__global__ __launch_bounds__(128) void conv_in_k(const float* __restrict__ x,
                                                 const float* __restrict__ wi,
                                                 const float* __restrict__ bi,
                                                 const float* __restrict__ pos,
                                                 _Float16* __restrict__ dst) {
    int xx = blockIdx.x, y = blockIdx.y, b = blockIdx.z, c = threadIdx.x;
    float v;
    if (c < 112) {
        v = bi[c];
        for (int ky = 0; ky < 3; ky++) {
            int iy = y + ky - 1;
            if (iy < 0 || iy >= GRID_N) continue;
            for (int kx = 0; kx < 3; kx++) {
                int ix = xx + kx - 1;
                if (ix < 0 || ix >= GRID_N) continue;
                v += x[b * 16384 + iy * GRID_N + ix] * wi[(c * 3 + ky) * 3 + kx];
            }
        }
    } else {
        v = pos[(c - 112) * 16384 + y * GRID_N + xx];
    }
    dst[(((size_t)(b * GRID_N + y) * GRID_N + xx) * CH) + c] = (_Float16)v;
}

// ---------------------------------------------------------------------------
// Implicit-GEMM 3x3 conv, 128->128 ch, WMMA f32_16x16x32_f16.
// One wave = 16 pixels x all 128 out channels (8 C tiles).
// stride: output stride (1 or 2). ups: input is 2x nearest-upsampled virtually.
// ---------------------------------------------------------------------------
__global__ __launch_bounds__(128) void conv3x3_wmma_k(const _Float16* __restrict__ src,
                                                      const _Float16* __restrict__ wpk,
                                                      const float* __restrict__ bias,
                                                      _Float16* __restrict__ dst,
                                                      float* __restrict__ dstf,
                                                      int Hs, int Ws, int Ho, int Wo,
                                                      int stride, int ups, int lrelu) {
    int wave = threadIdx.x >> 5, lane = threadIdx.x & 31;
    int tx = (blockIdx.x * 4 + wave) * 16;
    if (tx >= Wo) return;                       // wave-uniform
    int oy = blockIdx.y, b = blockIdx.z;
    int n = lane & 15, hi = lane >> 4;
    int ox = tx + n;
    int Hv = Hs << ups, Wv = Ws << ups;

    v8f acc[8];
#pragma unroll
    for (int t = 0; t < 8; t++)
#pragma unroll
        for (int r = 0; r < 8; r++) acc[t][r] = 0.f;

    for (int dy = 0; dy < 3; dy++) {
        int iy = oy * stride + dy - 1;
        bool yok = (iy >= 0) && (iy < Hv);
        int iyy = iy >> ups;
        for (int dx = 0; dx < 3; dx++) {
            int ix = ox * stride + dx - 1;
            bool ok = yok && (ix >= 0) && (ix < Wv);
            int ixx = ix >> ups;
            const _Float16* sp =
                src + (((size_t)(b * Hs + iyy) * Ws + ixx) * CH + hi * 16);
            int tap = dy * 3 + dx;
            for (int kb = 0; kb < 4; kb++) {
                v16h bf;
                if (ok) {
                    bf = *(const v16h*)(sp + kb * 32);
                } else {
#pragma unroll
                    for (int q = 0; q < 16; q++) bf[q] = (_Float16)0.f;
                }
                const _Float16* wp0 = wpk + ((size_t)((tap * 4 + kb) * 8) * 32 + lane) * 16;
#pragma unroll
                for (int t = 0; t < 8; t++) {
                    v16h af = *(const v16h*)(wp0 + (size_t)t * 512);
                    acc[t] = __builtin_amdgcn_wmma_f32_16x16x32_f16(
                        false, af, false, bf, (short)0, acc[t], false, false);
                }
            }
        }
    }

    size_t obase = ((size_t)(b * Ho + oy) * Wo + ox) * CH;
#pragma unroll
    for (int t = 0; t < 8; t++) {
        int cb = t * 16 + hi * 8;
        v8h hv;
#pragma unroll
        for (int r = 0; r < 8; r++) {
            float v = acc[t][r] + bias[cb + r];
            if (lrelu) v = (v >= 0.f) ? v : 0.2f * v;
            hv[r] = (_Float16)v;
            if (dstf) dstf[obase + cb + r] = v;
        }
        *(v8h*)(dst + obase + cb) = hv;
    }
}

// ---------------------------------------------------------------------------
// select logits (L2-normalized 1x1 conv) + softmax over 16384 -> feat_map
// ---------------------------------------------------------------------------
__global__ __launch_bounds__(1024) void sel_softmax_k(const float* __restrict__ hf,
                                                      const float* __restrict__ wsel,
                                                      const float* __restrict__ bsel,
                                                      float* __restrict__ feat) {
    __shared__ float wls[CH];
    __shared__ float red[1024];
    int b = blockIdx.x, t = threadIdx.x;
    if (t < CH) wls[t] = wsel[t];
    __syncthreads();
    float lg[16];
    float mx = -1e30f;
    for (int k = 0; k < 16; k++) {
        int p = k * 1024 + t;
        const float* hp = hf + ((size_t)b * 16384 + p) * CH;
        float nn = 0.f, dot = 0.f;
        for (int c = 0; c < CH; c++) {
            float v = hp[c];
            nn += v * v;
            dot += wls[c] * v;
        }
        float l = bsel[0] + dot / (sqrtf(nn) + 1e-8f);
        lg[k] = l;
        mx = fmaxf(mx, l);
    }
    red[t] = mx; __syncthreads();
    for (int s = 512; s > 0; s >>= 1) {
        if (t < s) red[t] = fmaxf(red[t], red[t + s]);
        __syncthreads();
    }
    float bmax = red[0]; __syncthreads();
    float ls = 0.f;
    for (int k = 0; k < 16; k++) { lg[k] = expf(lg[k] - bmax); ls += lg[k]; }
    red[t] = ls; __syncthreads();
    for (int s = 512; s > 0; s >>= 1) {
        if (t < s) red[t] += red[t + s];
        __syncthreads();
    }
    float tot = red[0]; __syncthreads();
    for (int k = 0; k < 16; k++) feat[(size_t)b * 16384 + k * 1024 + t] = lg[k] / tot;
}

// ---------------------------------------------------------------------------
// iterative top-16 argmax (ties -> lowest index, matches jax.lax.top_k)
// ---------------------------------------------------------------------------
__global__ __launch_bounds__(256) void topk_k(const float* __restrict__ feat,
                                              int* __restrict__ idx) {
    __shared__ unsigned char taken[16384];
    __shared__ float bv[256];
    __shared__ int bi_[256];
    int b = blockIdx.x, t = threadIdx.x;
    for (int p = t; p < 16384; p += 256) taken[p] = 0;
    __syncthreads();
    for (int it = 0; it < NATOM; it++) {
        float best = -1e30f;
        int bp = 1 << 30;
        for (int p = t; p < 16384; p += 256) {
            if (!taken[p]) {
                float v = feat[(size_t)b * 16384 + p];
                if (v > best || (v == best && p < bp)) { best = v; bp = p; }
            }
        }
        bv[t] = best; bi_[t] = bp; __syncthreads();
        for (int s = 128; s > 0; s >>= 1) {
            if (t < s) {
                if (bv[t + s] > bv[t] || (bv[t + s] == bv[t] && bi_[t + s] < bi_[t])) {
                    bv[t] = bv[t + s]; bi_[t] = bi_[t + s];
                }
            }
            __syncthreads();
        }
        if (t == 0) { idx[b * NATOM + it] = bi_[0]; taken[bi_[0]] = 1; }
        __syncthreads();
    }
}

// ---------------------------------------------------------------------------
// latents: 1x1 conv (wval) at the 16 selected pixels
// ---------------------------------------------------------------------------
__global__ __launch_bounds__(128) void latents_k(const float* __restrict__ hf,
                                                 const float* __restrict__ wval,
                                                 const float* __restrict__ bval,
                                                 const int* __restrict__ idx,
                                                 float* __restrict__ lat,
                                                 float* __restrict__ zb) {
    int blk = blockIdx.x;                // b*16 + atom
    int b = blk >> 4, j = threadIdx.x;
    int p = idx[blk];
    const float* hp = hf + ((size_t)b * 16384 + p) * CH;
    float s = bval[j];
    for (int i = 0; i < CH; i++) s += wval[j * CH + i] * hp[i];
    lat[blk * CH + j] = s;
    zb[blk * CH + j]  = s;
}

// ---------------------------------------------------------------------------
// per-atom decoder: z->512 linear, 3x (repeat2 + conv1d k3 + lrelu), heads
// ---------------------------------------------------------------------------
__global__ __launch_bounds__(128) void decoder_k(const float* __restrict__ z,
                                                 const float* __restrict__ wup, const float* __restrict__ bup,
                                                 const float* __restrict__ wu1, const float* __restrict__ bu1,
                                                 const float* __restrict__ wu2, const float* __restrict__ bu2,
                                                 const float* __restrict__ wu3, const float* __restrict__ bu3,
                                                 const float* __restrict__ wamp, const float* __restrict__ bamp,
                                                 const float* __restrict__ wfr,  const float* __restrict__ bfr,
                                                 const float* __restrict__ wnz,  const float* __restrict__ bnz,
                                                 float* __restrict__ ampo, float* __restrict__ fro,
                                                 float* __restrict__ mgo) {
    __shared__ float zs[CH];
    __shared__ float A[CH * 32];
    __shared__ float Bf[CH * 32];
    int a = blockIdx.x, c = threadIdx.x;
    zs[c] = z[a * CH + c];
    __syncthreads();
    for (int l = 0; l < 4; l++) {
        float s = bup[c * 4 + l];
        for (int i = 0; i < CH; i++) s += zs[i] * wup[i * 512 + c * 4 + l];
        A[c * 32 + l] = s;
    }
    __syncthreads();
    int Lp = 4;
    for (int r = 0; r < 3; r++) {
        const float* w  = (r == 0) ? wu1 : ((r == 1) ? wu2 : wu3);
        const float* bb = (r == 0) ? bu1 : ((r == 1) ? bu2 : bu3);
        int Lu = Lp * 2;
        for (int l = 0; l < Lu; l++) {
            float s = bb[c];
            for (int i = 0; i < CH; i++) {
                const float* wc = &w[(c * CH + i) * 3];
                if (l > 0)      s += wc[0] * A[i * 32 + ((l - 1) >> 1)];
                                s += wc[1] * A[i * 32 + (l >> 1)];
                if (l + 1 < Lu) s += wc[2] * A[i * 32 + ((l + 1) >> 1)];
            }
            Bf[c * 32 + l] = (s >= 0.f) ? s : 0.2f * s;
        }
        __syncthreads();
        for (int l = 0; l < Lu; l++) A[c * 32 + l] = Bf[c * 32 + l];
        __syncthreads();
        Lp = Lu;
    }
    // heads on u = A (128 x 32)
    for (int l = 0; l < 32; l++) {
        float sa = bamp[c], sf = bfr[c];
        for (int i = 0; i < CH; i++) {
            float u = A[i * 32 + l];
            sa += wamp[c * CH + i] * u;
            sf += wfr[c * CH + i] * u;
        }
        ampo[((size_t)a * CH + c) * 32 + l] = sa * sa;
        float sg = 1.f / (1.f + expf(-sf));
        fro[((size_t)a * CH + c) * 32 + l] = LOWF + sg * (1.f - LOWF);
    }
    for (int cc = c; cc < 129; cc += 128) {
        for (int l = 0; l < 32; l++) {
            float s = bnz[cc];
            for (int i = 0; i < CH; i++) s += wnz[cc * CH + i] * A[i * 32 + l];
            mgo[((size_t)a * 129 + cc) * 32 + l] = s * s;
        }
    }
}

// ---------------------------------------------------------------------------
// harmonic synth: phase = pi*cumsum(repeat(fr,256)); harm = sum_c sin*amp
// closed form: phase(j*256+r) = pi*(256*pre[j] + (r+1)*fr[j])
// ---------------------------------------------------------------------------
__global__ __launch_bounds__(256) void harm_k(const float* __restrict__ ampb,
                                              const float* __restrict__ frb,
                                              float* __restrict__ atoms) {
    __shared__ float pre[CH * 32];
    __shared__ float frv[CH * 32];
    __shared__ float amv[CH * 32];
    int a = blockIdx.x, t = threadIdx.x;
    if (t < CH) {
        float run = 0.f;
        for (int j = 0; j < 32; j++) {
            float f = frb[((size_t)a * CH + t) * 32 + j];
            pre[t * 32 + j] = run;
            run += f;
            frv[t * 32 + j] = f;
            amv[t * 32 + j] = ampb[((size_t)a * CH + t) * 32 + j];
        }
    }
    __syncthreads();
    int s = blockIdx.y * 256 + t;
    int j = s >> 8, r = s & 255;
    const float PI = 3.14159265358979323846f;
    float sum = 0.f;
    for (int c = 0; c < CH; c++) {
        float ph = PI * (256.f * pre[c * 32 + j] + (float)(r + 1) * frv[c * 32 + j]);
        sum += sinf(ph) * amv[c * 32 + j];
    }
    atoms[(size_t)a * ALEN + s] = sum;
}

// ---------------------------------------------------------------------------
// noise branch: per (atom, frame): DFT(256) * mags, iDFT(256); += into atoms.
// Twiddle table in LDS -> no transcendentals in inner loops.
// ---------------------------------------------------------------------------
__global__ __launch_bounds__(256) void noise_k(const float* __restrict__ noise,
                                               const float* __restrict__ mgs,
                                               float* __restrict__ atoms) {
    __shared__ float xn[256];
    __shared__ float tw[256];
    __shared__ float re[129];
    __shared__ float im[129];
    int a = blockIdx.x, fr = blockIdx.y, t = threadIdx.x;
    xn[t] = noise[((size_t)a * 32 + fr) * 256 + t];
    tw[t] = cosf((float)t * (6.283185307179586f / 256.f));
    __syncthreads();
    if (t < 129) {
        float r = 0.f, i = 0.f;
        for (int n = 0; n < 256; n++) {
            int m = (t * n) & 255;
            r += xn[n] * tw[m];
            i -= xn[n] * tw[(m + 192) & 255];   // sin(th) = cos(th - pi/2)
        }
        float mg = mgs[((size_t)a * 129 + t) * 32 + fr];
        re[t] = r * mg;
        im[t] = i * mg;
    }
    __syncthreads();
    int rr = t;
    float acc = re[0] + re[128] * ((rr & 1) ? -1.f : 1.f);
    for (int k = 1; k < 128; k++) {
        int m = (k * rr) & 255;
        acc += 2.f * (re[k] * tw[m] - im[k] * tw[(m + 192) & 255]);
    }
    atoms[(size_t)a * ALEN + fr * 256 + rr] += acc * (1.f / 256.f);
}

// ---------------------------------------------------------------------------
// deterministic overlap-add (gather form): out[p] = sum over atoms covering p
// ---------------------------------------------------------------------------
__global__ __launch_bounds__(256) void overlap_k(const float* __restrict__ atoms,
                                                 const int* __restrict__ idx,
                                                 float* __restrict__ out) {
    int g = blockIdx.x * 256 + threadIdx.x;       // 0 .. 2*32768-1
    int b = g >> 15, p = g & 32767;
    float s = 0.f;
    for (int a = 0; a < NATOM; a++) {
        int id = idx[b * NATOM + a];
        int d = p - id;
        if (d >= 0 && d < ALEN) s += atoms[((size_t)(b * NATOM + a)) * ALEN + d];
    }
    out[g] = s;
}

// ---------------------------------------------------------------------------
extern "C" void kernel_launch(void* const* d_in, const int* in_sizes, int n_in,
                              void* d_out, int out_size, void* d_ws, size_t ws_size,
                              hipStream_t stream) {
    (void)in_sizes; (void)n_in; (void)out_size; (void)ws_size;
    const float* x    = (const float*)d_in[0];
    const float* pos  = (const float*)d_in[1];
    const float* wi   = (const float*)d_in[2];
    const float* bi   = (const float*)d_in[3];
    const float* wp   = (const float*)d_in[4];
    const float* bp   = (const float*)d_in[5];
    const float* w1   = (const float*)d_in[6];
    const float* b1   = (const float*)d_in[7];
    const float* w2   = (const float*)d_in[8];
    const float* b2   = (const float*)d_in[9];
    const float* w3   = (const float*)d_in[10];
    const float* b3   = (const float*)d_in[11];
    const float* w4   = (const float*)d_in[12];
    const float* b4   = (const float*)d_in[13];
    const float* wsb  = (const float*)d_in[14];
    const float* bsb  = (const float*)d_in[15];
    const float* wsel = (const float*)d_in[16];
    const float* bsel = (const float*)d_in[17];
    const float* wval = (const float*)d_in[18];
    const float* bval = (const float*)d_in[19];
    const float* wup  = (const float*)d_in[20];
    const float* bup  = (const float*)d_in[21];
    const float* wu1  = (const float*)d_in[22];
    const float* bu1  = (const float*)d_in[23];
    const float* wu2  = (const float*)d_in[24];
    const float* bu2  = (const float*)d_in[25];
    const float* wu3  = (const float*)d_in[26];
    const float* bu3  = (const float*)d_in[27];
    const float* wamp = (const float*)d_in[28];
    const float* bamp = (const float*)d_in[29];
    const float* wfr  = (const float*)d_in[30];
    const float* bfr  = (const float*)d_in[31];
    const float* wnz  = (const float*)d_in[32];
    const float* bnz  = (const float*)d_in[33];
    const float* nz   = (const float*)d_in[34];

    // workspace layout (~38 MB total)
    char* w = (char*)d_ws;
    size_t off = 0;
    auto alloc = [&](size_t bytes) -> char* {
        char* p = w + off;
        off += (bytes + 255) & ~(size_t)255;
        return p;
    };
    _Float16* wpk[6];
    for (int i = 0; i < 6; i++) wpk[i] = (_Float16*)alloc(9 * 4 * 8 * 32 * 16 * 2);
    _Float16* actA = (_Float16*)alloc((size_t)NB * 128 * 128 * CH * 2);
    _Float16* actB = (_Float16*)alloc((size_t)NB * 128 * 128 * CH * 2);
    float* hf    = (float*)alloc((size_t)NB * 16384 * CH * 4);
    int*   idxb  = (int*)alloc(NB * NATOM * 4);
    float* zb    = (float*)alloc((size_t)NB * NATOM * CH * 4);
    float* ampb  = (float*)alloc((size_t)NB * NATOM * CH * 32 * 4);
    float* frb   = (float*)alloc((size_t)NB * NATOM * CH * 32 * 4);
    float* mgb   = (float*)alloc((size_t)NB * NATOM * 129 * 32 * 4);
    float* atoms = (float*)alloc((size_t)NB * NATOM * ALEN * 4);

    float* outp = (float*)d_out;          // (2,1,32768)
    float* lat  = outp + 65536;           // (2,16,128)
    float* feat = outp + 65536 + 4096;    // (2,128,128)

    // 1) pack conv weights into WMMA A-fragment layout
    const float* cw[6] = {wp, w1, w2, w3, w4, wsb};
    for (int i = 0; i < 6; i++)
        pack_weights_k<<<576, 256, 0, stream>>>(cw[i], wpk[i]);

    // 2) input conv + pos concat -> NHWC f16
    conv_in_k<<<dim3(128, 128, NB), 128, 0, stream>>>(x, wi, bi, pos, actA);

    // 3) conv stack (WMMA implicit GEMM)
    conv3x3_wmma_k<<<dim3(2, 128, NB), 128, 0, stream>>>(actA, wpk[0], bp, actB, (float*)nullptr, 128, 128, 128, 128, 1, 0, 0);
    conv3x3_wmma_k<<<dim3(1,  64, NB), 128, 0, stream>>>(actB, wpk[1], b1, actA, (float*)nullptr, 128, 128,  64,  64, 2, 0, 1);
    conv3x3_wmma_k<<<dim3(1,  32, NB), 128, 0, stream>>>(actA, wpk[2], b2, actB, (float*)nullptr,  64,  64,  32,  32, 2, 0, 1);
    conv3x3_wmma_k<<<dim3(1,  64, NB), 128, 0, stream>>>(actB, wpk[3], b3, actA, (float*)nullptr,  32,  32,  64,  64, 1, 1, 1);
    conv3x3_wmma_k<<<dim3(2, 128, NB), 128, 0, stream>>>(actA, wpk[4], b4, actB, (float*)nullptr,  64,  64, 128, 128, 1, 1, 1);
    conv3x3_wmma_k<<<dim3(2, 128, NB), 128, 0, stream>>>(actB, wpk[5], bsb, actA, hf,            128, 128, 128, 128, 1, 0, 0);

    // 4) selection softmax -> feat_map output
    sel_softmax_k<<<NB, 1024, 0, stream>>>(hf, wsel, bsel, feat);

    // 5) deterministic top-16
    topk_k<<<NB, 256, 0, stream>>>(feat, idxb);

    // 6) latents output (+ decoder input)
    latents_k<<<NB * NATOM, 128, 0, stream>>>(hf, wval, bval, idxb, lat, zb);

    // 7) per-atom decoder + heads
    decoder_k<<<NB * NATOM, 128, 0, stream>>>(zb, wup, bup, wu1, bu1, wu2, bu2, wu3, bu3,
                                              wamp, bamp, wfr, bfr, wnz, bnz,
                                              ampb, frb, mgb);

    // 8) harmonic synthesis (writes atoms), then noise branch (accumulates)
    harm_k<<<dim3(NB * NATOM, ALEN / 256), 256, 0, stream>>>(ampb, frb, atoms);
    noise_k<<<dim3(NB * NATOM, 32), 256, 0, stream>>>(nz, mgb, atoms);

    // 9) deterministic overlap-add -> output
    overlap_k<<<(NB * NSAMP) / 256, 256, 0, stream>>>(atoms, idxb, outp);
}
